// UnifiedTransformer_25769803799
// MI455X (gfx1250) — matrix operand order repaired and checked by reading, not compiled, not run on hardware
//
#include <hip/hip_runtime.h>
#include <cstdint>

typedef unsigned short u16;
typedef __attribute__((ext_vector_type(16))) __bf16 v16bf;
typedef __attribute__((ext_vector_type(8)))  float  v8f;
typedef int v4i __attribute__((vector_size(16)));

union Frag { uint4 u4[2]; v16bf v; };

// ---- CDNA5 async global->LDS copy (ASYNCcnt path), with safe fallback ------

#if defined(__HIP_DEVICE_COMPILE__) && __has_builtin(__builtin_amdgcn_global_load_async_to_lds_b128)
#define HAS_ASYNC 1
#else
#define HAS_ASYNC 0
#endif

typedef __attribute__((address_space(1))) v4i* gptr_t;   // global int4*
typedef __attribute__((address_space(3))) v4i* lptr_t;   // LDS int4*

__device__ __forceinline__ void cp16(const u16* g, u16* l) {
#if HAS_ASYNC
  // flat LDS pointers carry the LDS byte offset in addr[31:0] (ISA 10.2),
  // so truncating to a 32-bit AS3 pointer is valid.
  __builtin_amdgcn_global_load_async_to_lds_b128(
      (gptr_t)(uintptr_t)g,
      (lptr_t)(unsigned)(uintptr_t)l,
      0, 0);
#else
  *(uint4*)l = *(const uint4*)g;
#endif
}

__device__ __forceinline__ void wait_async() {
#if HAS_ASYNC
#if __has_builtin(__builtin_amdgcn_s_wait_asynccnt)
  __builtin_amdgcn_s_wait_asynccnt(0);
#else
  asm volatile("s_wait_asynccnt 0x0" ::: "memory");
#endif
#endif
}

// ---- helpers ---------------------------------------------------------------

__device__ __forceinline__ u16 f2bf(float f) {
  unsigned u = __float_as_uint(f);
  unsigned r = (u + 0x7fffu + ((u >> 16) & 1u)) >> 16;
  return (u16)r;
}

__device__ __forceinline__ float bsum(float v, float* sh) {
  int t = threadIdx.x;
  sh[t] = v; __syncthreads();
  for (int s = 128; s >= 1; s >>= 1) { if (t < s) sh[t] += sh[t + s]; __syncthreads(); }
  float r = sh[0]; __syncthreads(); return r;
}

__device__ __forceinline__ float bmax(float v, float* sh) {
  int t = threadIdx.x;
  sh[t] = v; __syncthreads();
  for (int s = 128; s >= 1; s >>= 1) { if (t < s) sh[t] = fmaxf(sh[t], sh[t + s]); __syncthreads(); }
  float r = sh[0]; __syncthreads(); return r;
}

// ---- f32 -> bf16 convert ---------------------------------------------------

__global__ void __launch_bounds__(256) k_cvt(const float* __restrict__ s,
                                             u16* __restrict__ d, long long n) {
  long long stride = (long long)gridDim.x * 256;
  for (long long i = (long long)blockIdx.x * 256 + threadIdx.x; i < n; i += stride)
    d[i] = f2bf(s[i]);
}

// ---- embedding + LayerNorm -------------------------------------------------

__global__ void __launch_bounds__(256) k_embed(
    const int* __restrict__ tok, const int* __restrict__ seg, const int* __restrict__ pos,
    const float* __restrict__ tE, const float* __restrict__ sE, const float* __restrict__ pE,
    const float* __restrict__ g, const float* __restrict__ be,
    float* __restrict__ of, u16* __restrict__ ob) {
  __shared__ float sh[256];
  int t = blockIdx.x, i = threadIdx.x;
  int tk = tok[t], sg = seg[t], ps = pos[t];
  float v[3];
#pragma unroll
  for (int e = 0; e < 3; ++e) {
    int d = i + e * 256;
    v[e] = tE[(size_t)tk * 768 + d] * 27.712812921102035f
         + sE[(size_t)sg * 768 + d] + pE[(size_t)ps * 768 + d];
  }
  float m = bsum(v[0] + v[1] + v[2], sh) * (1.f / 768.f);
  float var = bsum((v[0]-m)*(v[0]-m) + (v[1]-m)*(v[1]-m) + (v[2]-m)*(v[2]-m), sh) * (1.f / 768.f);
  float rs = rsqrtf(var + 1e-5f);
#pragma unroll
  for (int e = 0; e < 3; ++e) {
    int d = i + e * 256;
    float o = (v[e] - m) * rs * g[d] + be[d];
    of[(size_t)t * 768 + d] = o;
    ob[(size_t)t * 768 + d] = f2bf(o);
  }
}

// ---- residual add + LayerNorm ---------------------------------------------

__global__ void __launch_bounds__(256) k_add_ln(
    const float* __restrict__ x, const float* __restrict__ dl,
    const float* __restrict__ g, const float* __restrict__ be,
    float* __restrict__ of, u16* __restrict__ ob) {
  __shared__ float sh[256];
  int t = blockIdx.x, i = threadIdx.x;
  float v[3];
#pragma unroll
  for (int e = 0; e < 3; ++e) {
    int d = i + e * 256;
    v[e] = x[(size_t)t * 768 + d] + dl[(size_t)t * 768 + d];
  }
  float m = bsum(v[0] + v[1] + v[2], sh) * (1.f / 768.f);
  float var = bsum((v[0]-m)*(v[0]-m) + (v[1]-m)*(v[1]-m) + (v[2]-m)*(v[2]-m), sh) * (1.f / 768.f);
  float rs = rsqrtf(var + 1e-5f);
#pragma unroll
  for (int e = 0; e < 3; ++e) {
    int d = i + e * 256;
    float o = (v[e] - m) * rs * g[d] + be[d];
    of[(size_t)t * 768 + d] = o;
    ob[(size_t)t * 768 + d] = f2bf(o);
  }
}

// ---- qkv -> per-head q (scaled), k, v-transposed ---------------------------

__global__ void __launch_bounds__(256) k_to_heads(
    const float* __restrict__ qkv, u16* __restrict__ q, u16* __restrict__ k,
    u16* __restrict__ vt) {
  long long stride = (long long)gridDim.x * 256;
  for (long long idx = (long long)blockIdx.x * 256 + threadIdx.x;
       idx < (long long)2048 * 768; idx += stride) {
    int t = (int)(idx / 768), d = (int)(idx % 768);
    int s = t >> 3, b = t & 7;       // token = s*B + b, B=8
    int h = d >> 6, hd = d & 63;     // HD=64
    int bh = b * 12 + h;
    const float* base = qkv + (size_t)t * 2304;
    q [((size_t)bh * 256 + s) * 64 + hd] = f2bf(base[d] * 0.125f);     // HD^-0.5
    k [((size_t)bh * 256 + s) * 64 + hd] = f2bf(base[768 + d]);
    vt[((size_t)bh * 64 + hd) * 256 + s] = f2bf(base[1536 + d]);
  }
}

// ---- merge heads: [B,H,S,HD] f32 -> [S*B, D] bf16 --------------------------

__global__ void __launch_bounds__(256) k_merge(
    const float* __restrict__ a, u16* __restrict__ o) {
  long long stride = (long long)gridDim.x * 256;
  for (long long idx = (long long)blockIdx.x * 256 + threadIdx.x;
       idx < (long long)2048 * 768; idx += stride) {
    int t = (int)(idx / 768), d = (int)(idx % 768);
    int s = t >> 3, b = t & 7;
    int h = d >> 6, hd = d & 63;
    o[idx] = f2bf(a[((size_t)(b * 12 + h) * 256 + s) * 64 + hd]);
  }
}

// ---- masked softmax over rows of length 256 --------------------------------

__global__ void __launch_bounds__(256) k_softmax(
    const float* __restrict__ sc, const float* __restrict__ mask,
    u16* __restrict__ ob) {
  __shared__ float sh[256];
  int r = blockIdx.x;            // (b*H + h)*S + q
  int i = threadIdx.x;
  int q = r & 255;
  int b = r / (12 * 256);
  float v = sc[(size_t)r * 256 + i] + mask[((size_t)b * 256 + q) * 256 + i];
  float mx = bmax(v, sh);
  float e = __expf(v - mx);
  float s = bsum(e, sh);
  ob[(size_t)r * 256 + i] = f2bf(e / s);
}

// ---- WMMA GEMM: C[M,N] = A[M,K] * B[N,K]^T (+bias, opt relu), batched ------
// block tile 128x64, 8 waves, wave tile 32x32 (2x2 WMMA bf16 16x16x32 frags)
// double-buffered LDS, async global->LDS staging (ASYNCcnt)

__device__ __forceinline__ void stage_tile(
    const u16* __restrict__ A, const u16* __restrict__ B, int K,
    int m0, int n0, int k0, u16* a_lds, u16* b_lds, int tid) {
  // A: 128 rows x 32 halves; 2 chunks of 16B per thread (contiguous 32B)
  int ar = tid >> 1, ap = (tid & 1) * 16;
  const u16* ga = A + (size_t)(m0 + ar) * K + k0 + ap;
  u16* la = a_lds + ar * 40 + ap;
  cp16(ga, la);
  cp16(ga + 8, la + 8);
  // B: 64 rows x 32 halves; 1 chunk per thread
  int br = tid >> 2, bp = (tid & 3) * 8;
  cp16(B + (size_t)(n0 + br) * K + k0 + bp, b_lds + br * 40 + bp);
}

__global__ void __launch_bounds__(256) k_gemm_nt(
    const u16* __restrict__ A, const u16* __restrict__ B,
    const float* __restrict__ bias, float* __restrict__ C, u16* __restrict__ Cbf,
    int M, int N, int K, long long sA, long long sB, long long sC, int relu) {
  __shared__ alignas(16) u16 sAl[2][128 * 40];
  __shared__ alignas(16) u16 sBl[2][64 * 40];
  A += (long long)blockIdx.z * sA;
  B += (long long)blockIdx.z * sB;
  const long long cofs = (long long)blockIdx.z * sC;
  const int m0 = blockIdx.y * 128;
  const int n0 = blockIdx.x * 64;
  const int tid = threadIdx.x;
  const int lane = tid & 31, wv = tid >> 5;
  const int wm = (wv >> 1) << 5;   // 0,32,64,96
  const int wn = (wv & 1) << 5;    // 0,32
  const int lh = lane & 15, hh = lane >> 4;

  v8f acc[2][2] = {};

  stage_tile(A, B, K, m0, n0, 0, sAl[0], sBl[0], tid);
  wait_async();
  __syncthreads();

  int buf = 0;
  for (int k0 = 0; k0 < K; k0 += 32) {
    if (k0 + 32 < K)   // prefetch next K-tile into the other buffer
      stage_tile(A, B, K, m0, n0, k0 + 32, sAl[buf ^ 1], sBl[buf ^ 1], tid);

    const u16* ab = sAl[buf];
    const u16* bb = sBl[buf];
    Frag a[2], b[2];
#pragma unroll
    for (int i = 0; i < 2; ++i) {           // A: lane<16 -> K 0..7 & 16..23; lane>=16 -> +8
      int r = wm + i * 16 + lh;
      a[i].u4[0] = *(const uint4*)&ab[r * 40 + hh * 8];
      a[i].u4[1] = *(const uint4*)&ab[r * 40 + 16 + hh * 8];
    }
#pragma unroll
    for (int j = 0; j < 2; ++j) {           // B: lane<16 -> K 0..7 & 16..23; lane>=16 -> +8
      int n = wn + j * 16 + lh;
      b[j].u4[0] = *(const uint4*)&bb[n * 40 + hh * 8];
      b[j].u4[1] = *(const uint4*)&bb[n * 40 + 16 + hh * 8];
    }
#pragma unroll
    for (int i = 0; i < 2; ++i)
#pragma unroll
      for (int j = 0; j < 2; ++j)
        acc[i][j] = __builtin_amdgcn_wmma_f32_16x16x32_bf16(
            false, a[i].v, false, b[j].v, (short)0, acc[i][j], false, false);

    wait_async();
    __syncthreads();
    buf ^= 1;
  }

#pragma unroll
  for (int i = 0; i < 2; ++i) {
    int mr0 = m0 + wm + i * 16 + hh * 8;    // C/D: vgpr r -> row r (+8 for hi lanes)
#pragma unroll
    for (int j = 0; j < 2; ++j) {
      int nc = n0 + wn + j * 16 + lh;
      float bv = bias ? bias[nc] : 0.f;
#pragma unroll
      for (int r = 0; r < 8; ++r) {
        float v = acc[i][j][r] + bv;
        if (relu) v = v > 0.f ? v : 0.f;
        size_t idx = (size_t)(cofs + (long long)(mr0 + r) * N + nc);
        if (C)   C[idx] = v;
        if (Cbf) Cbf[idx] = f2bf(v);
      }
    }
  }
}

// ---- host side -------------------------------------------------------------

extern "C" void kernel_launch(void* const* d_in, const int* in_sizes, int n_in,
                              void* d_out, int out_size, void* d_ws, size_t ws_size,
                              hipStream_t stream) {
  (void)in_sizes; (void)n_in; (void)out_size; (void)ws_size;
  const int Dm = 768, Ff = 3072, Ll = 6, Vv = 32000, Ss = 256, TDd = 2304;
  const int NT = 2048, BH = 96, HDd = 64;

  const int*   tok   = (const int*)d_in[0];
  const int*   seg   = (const int*)d_in[1];
  const int*   pos   = (const int*)d_in[2];
  const float* amask = (const float*)d_in[3];
  const float* tokE  = (const float*)d_in[5];
  const float* segE  = (const float*)d_in[6];
  const float* posE  = (const float*)d_in[7];
  const float* lnEg  = (const float*)d_in[8];
  const float* lnEb  = (const float*)d_in[9];
  const float* Win   = (const float*)d_in[10];
  const float* bin   = (const float*)d_in[11];
  const float* Wout  = (const float*)d_in[12];
  const float* bout  = (const float*)d_in[13];
  const float* ln1g  = (const float*)d_in[14];
  const float* ln1b  = (const float*)d_in[15];
  const float* Wff1  = (const float*)d_in[16];
  const float* bff1  = (const float*)d_in[17];
  const float* Wff2  = (const float*)d_in[18];
  const float* bff2  = (const float*)d_in[19];
  const float* ln2g  = (const float*)d_in[20];
  const float* ln2b  = (const float*)d_in[21];
  const float* headw = (const float*)d_in[22];

  char* ws = (char*)d_ws;
  size_t off = 0;
  auto alloc = [&](size_t bytes) -> void* {
    void* p = ws + off;
    off += (bytes + 255) & ~(size_t)255;
    return p;
  };

  u16* wWin  = (u16*)alloc((size_t)Ll * TDd * Dm * 2);
  u16* wWout = (u16*)alloc((size_t)Ll * Dm * Dm * 2);
  u16* wWff1 = (u16*)alloc((size_t)Ll * Ff * Dm * 2);
  u16* wWff2 = (u16*)alloc((size_t)Ll * Dm * Ff * 2);
  u16* wHead = (u16*)alloc((size_t)Vv * Dm * 2);

  float* xf     = (float*)alloc((size_t)NT * Dm * 4);
  u16*   xbf    = (u16*)  alloc((size_t)NT * Dm * 2);
  float* qkvf   = (float*)alloc((size_t)NT * TDd * 4);
  u16*   qbf    = (u16*)  alloc((size_t)BH * Ss * HDd * 2);
  u16*   kbf    = (u16*)  alloc((size_t)BH * Ss * HDd * 2);
  u16*   vtbf   = (u16*)  alloc((size_t)BH * HDd * Ss * 2);
  float* scoref = (float*)alloc((size_t)BH * Ss * Ss * 4);
  u16*   attnbf = (u16*)  alloc((size_t)BH * Ss * Ss * 2);
  float* aoutf  = (float*)alloc((size_t)BH * Ss * HDd * 4);
  u16*   mrgbf  = (u16*)  alloc((size_t)NT * Dm * 2);
  float* aprojf = (float*)alloc((size_t)NT * Dm * 4);
  float* x2f    = (float*)alloc((size_t)NT * Dm * 4);
  u16*   x2bf   = (u16*)  alloc((size_t)NT * Dm * 2);
  u16*   hbf    = (u16*)  alloc((size_t)NT * Ff * 2);
  float* ffnf   = (float*)alloc((size_t)NT * Dm * 4);

  // convert all weights to bf16 once per launch
  k_cvt<<<4096, 256, 0, stream>>>(Win,  wWin,  (long long)Ll * TDd * Dm);
  k_cvt<<<4096, 256, 0, stream>>>(Wout, wWout, (long long)Ll * Dm * Dm);
  k_cvt<<<4096, 256, 0, stream>>>(Wff1, wWff1, (long long)Ll * Ff * Dm);
  k_cvt<<<4096, 256, 0, stream>>>(Wff2, wWff2, (long long)Ll * Dm * Ff);
  k_cvt<<<4096, 256, 0, stream>>>(headw, wHead, (long long)Vv * Dm);

  k_embed<<<NT, 256, 0, stream>>>(tok, seg, pos, tokE, segE, posE, lnEg, lnEb, xf, xbf);

  for (int l = 0; l < Ll; ++l) {
    // QKV projection: [NT,D] x [3D,D]^T
    k_gemm_nt<<<dim3(TDd / 64, NT / 128, 1), 256, 0, stream>>>(
        xbf, wWin + (size_t)l * TDd * Dm, bin + (size_t)l * TDd, qkvf, nullptr,
        NT, TDd, Dm, 0, 0, 0, 0);
    k_to_heads<<<2048, 256, 0, stream>>>(qkvf, qbf, kbf, vtbf);
    // scores = q k^T  (batched over B*H)
    k_gemm_nt<<<dim3(Ss / 64, Ss / 128, BH), 256, 0, stream>>>(
        qbf, kbf, nullptr, scoref, nullptr, Ss, Ss, HDd,
        (long long)Ss * HDd, (long long)Ss * HDd, (long long)Ss * Ss, 0);
    k_softmax<<<BH * Ss, 256, 0, stream>>>(scoref, amask, attnbf);
    // out = attn v   (B stored [N=HD, K=S] = v^T)
    k_gemm_nt<<<dim3(HDd / 64, Ss / 128, BH), 256, 0, stream>>>(
        attnbf, vtbf, nullptr, aoutf, nullptr, Ss, HDd, Ss,
        (long long)Ss * Ss, (long long)HDd * Ss, (long long)Ss * HDd, 0);
    k_merge<<<2048, 256, 0, stream>>>(aoutf, mrgbf);
    // output projection
    k_gemm_nt<<<dim3(Dm / 64, NT / 128, 1), 256, 0, stream>>>(
        mrgbf, wWout + (size_t)l * Dm * Dm, bout + (size_t)l * Dm, aprojf, nullptr,
        NT, Dm, Dm, 0, 0, 0, 0);
    k_add_ln<<<NT, 256, 0, stream>>>(xf, aprojf, ln1g + (size_t)l * Dm,
                                     ln1b + (size_t)l * Dm, x2f, x2bf);
    // FFN1 + ReLU, bf16 output fused into epilogue (no extra convert pass)
    k_gemm_nt<<<dim3(Ff / 64, NT / 128, 1), 256, 0, stream>>>(
        x2bf, wWff1 + (size_t)l * Ff * Dm, bff1 + (size_t)l * Ff, nullptr, hbf,
        NT, Ff, Dm, 0, 0, 0, 1);
    k_gemm_nt<<<dim3(Dm / 64, NT / 128, 1), 256, 0, stream>>>(
        hbf, wWff2 + (size_t)l * Dm * Ff, bff2 + (size_t)l * Dm, ffnf, nullptr,
        NT, Dm, Ff, 0, 0, 0, 0);
    k_add_ln<<<NT, 256, 0, stream>>>(x2f, ffnf, ln2g + (size_t)l * Dm,
                                     ln2b + (size_t)l * Dm, xf, xbf);
  }

  // LM head: [NT,D] x [V,D]^T -> logits f32 in d_out
  k_gemm_nt<<<dim3(Vv / 64, NT / 128, 1), 256, 0, stream>>>(
      xbf, wHead, nullptr, (float*)d_out, nullptr, NT, Vv, Dm, 0, 0, 0, 0);
}
